// _RPN_55482387530024
// MI455X (gfx1250) — matrix operand order repaired and checked by reading, not compiled, not run on hardware
//
#include <hip/hip_runtime.h>
#include <math.h>

typedef _Float16 half_t;
typedef __attribute__((ext_vector_type(16))) _Float16 v16h;
typedef __attribute__((ext_vector_type(8)))  _Float16 v8h;
typedef __attribute__((ext_vector_type(8)))  float    v8f;

#define BATCH 8
#define CIN   512
#define NOUT  512
#define HF    63
#define WFT   63
#define PIX   (HF*WFT)        // 3969
#define KDIM  (CIN*9)         // 4608
#define MROWS (BATCH*PIX)     // 31752
#define A9    (PIX*9)         // 35721
#define POST  2000
#define NMSCH 35              // 35*1024 >= A9
#define KSTEPS (KDIM/32)      // 144

// ---------------------------------------------------------------- anchors
__device__ __forceinline__ void anchor_of(int i, float& x1, float& y1,
                                          float& x2, float& y2) {
  int p = i / 9, a = i - p * 9;
  int oy = p / WFT, ox = p - oy * WFT;
  const float ratios[3] = {0.5f, 1.0f, 2.0f};
  const float scales[3] = {8.0f, 16.0f, 32.0f};
  int ri = a / 3, si = a - ri * 3;
  float ws = rintf(sqrtf(256.0f / ratios[ri]));
  float hs = rintf(ws * ratios[ri]);
  float Wd = ws * scales[si], Hd = hs * scales[si];
  float sx = ox * 16.0f, sy = oy * 16.0f;
  x1 = sx + 7.5f - 0.5f * (Wd - 1.0f);
  y1 = sy + 7.5f - 0.5f * (Hd - 1.0f);
  x2 = sx + 7.5f + 0.5f * (Wd - 1.0f);
  y2 = sy + 7.5f + 0.5f * (Hd - 1.0f);
}

// ---------------------------------------------------------------- prep: wTT f16
// wTT[n][k'] with k' = tap*512 + c (tap = ky*3+kx): column-major weights so
// B fragments are contiguous 32B runs in global memory.
__global__ void prep_wtt(const float* __restrict__ w, half_t* __restrict__ wtt, int n) {
  int i = blockIdx.x * blockDim.x + threadIdx.x;   // i = nn*KDIM + k'
  if (i >= n) return;
  int nn = i / KDIM;
  int kp = i - nn * KDIM;
  int r  = kp >> 9;        // tap 0..8
  int c  = kp & 511;       // channel
  wtt[i] = (half_t)w[(size_t)nn * KDIM + c * 9 + r];
}

// ---------------------------------------------------------------- conv 3x3 WMMA
// Implicit GEMM: D[M=31752, N=512] = im2col(A)[M, 4608] x W[4608, 512]
// K reordered tap-major/channel-minor; block tile 128Mx128N; wave tile 32Mx64N.
// Double-buffered LDS staging, branchless clamped-safe im2col gather.
__global__ __launch_bounds__(256) void conv3x3_wmma(
    const float* __restrict__ basef, const half_t* __restrict__ wtt,
    const float* __restrict__ bias, float* __restrict__ xout)
{
  __shared__ half_t As[2][128][40];   // two 128x32 f16 tiles (padded rows)

  const int tid  = threadIdx.x;
  const int lane = tid & 31;
  const int wave = tid >> 5;
  const int wm = wave & 3;          // wave M index (0..3) -> 32 rows each
  const int wn = wave >> 2;         // wave N index (0..1) -> 64 cols each
  const int blockM = blockIdx.x;    // 0..248
  const int blockN = blockIdx.y;    // 0..3

  // A staging map: row = tid&127, channel half c0 = (tid>>7)*16
  const int arow = tid & 127;
  const int c0   = (tid >> 7) * 16;
  const int m    = blockM * 128 + arow;
  const bool mok = (m < MROWS);
  const int ab  = mok ? (m / PIX) : 0;
  const int ap  = mok ? (m - ab * PIX) : 0;
  const int aoy = ap / WFT;
  const int aox = ap - aoy * WFT;

  v8f acc[2][4] = {};

  const int arow0 = wm * 32 + (lane & 15);   // first A-frag row
  const int ks    = (lane >> 4) * 8;         // A frag chunk base (0 or 8)
  const int koffB = (lane >> 4) * 8;         // B frag chunk base (0 or 8)

  // B fragment base pointers (global, column-major weights)
  const half_t* bptr[4];
  #pragma unroll
  for (int t = 0; t < 4; ++t) {
    int n = blockN * 128 + wn * 64 + t * 16 + (lane & 15);
    bptr[t] = wtt + (size_t)n * KDIM + koffB;
  }

  // Stage one 128x32 A tile (step = r*16 + kc/32) into buffer `buf`.
  // Loads are unconditional from a clamped-safe address (always inside
  // base_feat), zero-select applied afterwards -> no exec-mask branches.
  auto stage = [&](int step, int buf) {
    int r  = step >> 4;
    int kc = (step & 15) * 32;
    int ky = r / 3;
    int kx = r - ky * 3;
    int iy = aoy + ky - 1;
    int ix = aox + kx - 1;
    bool valid = mok && (unsigned)iy < (unsigned)HF &&
                 (unsigned)ix < (unsigned)WFT;
    int iyc = valid ? iy : 0;
    int ixc = valid ? ix : 0;
    const float* bp = basef +
        ((size_t)(ab * CIN + c0 + kc) * HF + iyc) * WFT + ixc;
    float v[16];
    #pragma unroll
    for (int j = 0; j < 16; ++j)
      v[j] = bp[(size_t)j * PIX];
    v8h pk0, pk1;
    #pragma unroll
    for (int j = 0; j < 8; ++j) {
      pk0[j] = (half_t)(valid ? v[j] : 0.0f);
      pk1[j] = (half_t)(valid ? v[8 + j] : 0.0f);
    }
    *(v8h*)&As[buf][arow][c0]     = pk0;
    *(v8h*)&As[buf][arow][c0 + 8] = pk1;
  };

  stage(0, 0);
  __syncthreads();

  for (int s = 0; s < KSTEPS; ++s) {
    const int nb = s & 1;
    if (s + 1 < KSTEPS) stage(s + 1, nb ^ 1);

    const int k0 = s * 32;

    // ---- A fragments (two 16x32 f16 tiles, documented lane layout)
    v8h a0lo = *(const v8h*)&As[nb][arow0][ks];
    v8h a0hi = *(const v8h*)&As[nb][arow0][16 + ks];
    v8h a1lo = *(const v8h*)&As[nb][arow0 + 16][ks];
    v8h a1hi = *(const v8h*)&As[nb][arow0 + 16][16 + ks];
    v16h af0, af1;
    #pragma unroll
    for (int q = 0; q < 8; ++q) {
      af0[q] = a0lo[q]; af0[8 + q] = a0hi[q];
      af1[q] = a1lo[q]; af1[8 + q] = a1hi[q];
    }

    // ---- B fragments straight from global (L2-resident weights)
    #pragma unroll
    for (int t = 0; t < 4; ++t) {
      v8h blo = *(const v8h*)(bptr[t] + k0);
      v8h bhi = *(const v8h*)(bptr[t] + k0 + 16);
      v16h bf;
      #pragma unroll
      for (int q = 0; q < 8; ++q) { bf[q] = blo[q]; bf[8 + q] = bhi[q]; }
      acc[0][t] = __builtin_amdgcn_wmma_f32_16x16x32_f16(
          false, af0, false, bf, (short)0, acc[0][t], false, false);
      acc[1][t] = __builtin_amdgcn_wmma_f32_16x16x32_f16(
          false, af1, false, bf, (short)0, acc[1][t], false, false);
    }
    __syncthreads();
  }

  // ---- epilogue: bias + relu, scatter to NCHW x
  const int ncl = lane & 15;
  const int hi8 = (lane >> 4) * 8;
  #pragma unroll
  for (int t = 0; t < 4; ++t) {
    const int n  = blockN * 128 + wn * 64 + t * 16 + ncl;
    const float bv = bias[n];
    #pragma unroll
    for (int s = 0; s < 2; ++s) {
      const int mbase = blockM * 128 + wm * 32 + s * 16 + hi8;
      #pragma unroll
      for (int rr = 0; rr < 8; ++rr) {
        int mm = mbase + rr;
        if (mm < MROWS) {
          int b = mm / PIX, p = mm - b * PIX;
          float val = acc[s][t][rr] + bv;
          xout[(size_t)(b * NOUT + n) * PIX + p] = val > 0.f ? val : 0.f;
        }
      }
    }
  }
}

// ---------------------------------------------------------------- 1x1 heads
__global__ __launch_bounds__(256) void head1x1(
    const float* __restrict__ x,
    const float* __restrict__ cls_w, const float* __restrict__ cls_b,
    const float* __restrict__ bbox_w, const float* __restrict__ bbox_b,
    float* __restrict__ cls_score, float* __restrict__ bbox_pred)
{
  int gid = blockIdx.x * 256 + threadIdx.x;
  int bp = gid % MROWS;
  int oo = gid / MROWS;
  if (oo >= 54) return;
  int b = bp / PIX, p = bp - b * PIX;
  const float* wrow = (oo < 18) ? (cls_w + (size_t)oo * CIN)
                                : (bbox_w + (size_t)(oo - 18) * CIN);
  const float* xp = x + (size_t)b * CIN * PIX + p;
  float acc = 0.f;
  #pragma unroll 4
  for (int c = 0; c < CIN; ++c) {
    if ((c & 127) == 0 && c + 128 < CIN)
      __builtin_prefetch(xp + (size_t)(c + 128) * PIX, 0, 1);
    acc += xp[(size_t)c * PIX] * wrow[c];
  }
  if (oo < 18)
    cls_score[((size_t)b * 18 + oo) * PIX + p] = acc + cls_b[oo];
  else
    bbox_pred[((size_t)b * 36 + (oo - 18)) * PIX + p] = acc + bbox_b[oo - 18];
}

// ---------------------------------------------------------------- proposals
__global__ void proposal_decode(
    const float* __restrict__ cls_score, const float* __restrict__ bbox_pred,
    const float* __restrict__ im_info,
    float* __restrict__ scores, float* __restrict__ props)
{
  int gid = blockIdx.x * blockDim.x + threadIdx.x;
  if (gid >= BATCH * A9) return;
  int b = gid / A9, i = gid - b * A9;
  int p = i / 9, a = i - p * 9;
  float x1, y1, x2, y2;
  anchor_of(i, x1, y1, x2, y2);
  float aw = x2 - x1 + 1.f, ah = y2 - y1 + 1.f;
  float ax = x1 + 0.5f * aw, ay = y1 + 0.5f * ah;
  size_t pb = ((size_t)b * 36 + a * 4) * PIX + p;
  float dx = bbox_pred[pb], dy = bbox_pred[pb + PIX];
  float dw = bbox_pred[pb + 2 * (size_t)PIX], dh = bbox_pred[pb + 3 * (size_t)PIX];
  float px = dx * aw + ax, py = dy * ah + ay;
  float pw = expf(dw) * aw, ph = expf(dh) * ah;
  float mx = im_info[b * 3 + 1] - 1.f, my = im_info[b * 3 + 0] - 1.f;
  float bx1 = fminf(fmaxf(px - 0.5f * pw, 0.f), mx);
  float by1 = fminf(fmaxf(py - 0.5f * ph, 0.f), my);
  float bx2 = fminf(fmaxf(px + 0.5f * pw, 0.f), mx);
  float by2 = fminf(fmaxf(py + 0.5f * ph, 0.f), my);
  float s0 = cls_score[((size_t)b * 18 + a) * PIX + p];
  float s1 = cls_score[((size_t)b * 18 + 9 + a) * PIX + p];
  scores[gid] = 1.f / (1.f + expf(s0 - s1));
  float* pr = props + (size_t)gid * 4;
  pr[0] = bx1; pr[1] = by1; pr[2] = bx2; pr[3] = by2;
}

// ---------------------------------------------------------------- greedy NMS
__global__ __launch_bounds__(1024) void nms_kernel(
    const float* __restrict__ scores, const float* __restrict__ props,
    float* __restrict__ out)
{
  __shared__ float sval[1024];
  __shared__ int   sidx[1024];
  __shared__ float sbox[5];
  const int b = blockIdx.x;
  const int tid = threadIdx.x;
  const float* sc_g = scores + (size_t)b * A9;
  const float* bx_g = props  + (size_t)b * A9 * 4;

  float sc[NMSCH];
  #pragma unroll
  for (int j = 0; j < NMSCH; ++j) {
    int idx = j * 1024 + tid;
    sc[j] = (idx < A9) ? sc_g[idx] : -3.4e38f;
  }

  for (int it = 0; it < POST; ++it) {
    float bv = -3.4e38f; int bi = 0;
    #pragma unroll
    for (int j = 0; j < NMSCH; ++j) {
      int idx = j * 1024 + tid;
      if (sc[j] > bv) { bv = sc[j]; bi = idx; }
    }
    sval[tid] = bv; sidx[tid] = bi;
    __syncthreads();
    for (int off = 512; off > 0; off >>= 1) {
      if (tid < off) {
        if (sval[tid + off] > sval[tid]) {
          sval[tid] = sval[tid + off]; sidx[tid] = sidx[tid + off];
        }
      }
      __syncthreads();
    }
    if (tid == 0) {
      int j = sidx[0];
      bool ok = sval[0] > -3.0e38f;
      const float* bb = bx_g + (size_t)j * 4;
      float b0 = bb[0], b1 = bb[1], b2 = bb[2], b3 = bb[3];
      float* o = out + ((size_t)b * POST + it) * 5;
      o[0] = (float)b;
      o[1] = ok ? b0 : 0.f; o[2] = ok ? b1 : 0.f;
      o[3] = ok ? b2 : 0.f; o[4] = ok ? b3 : 0.f;
      sbox[0] = b0; sbox[1] = b1; sbox[2] = b2; sbox[3] = b3;
      sbox[4] = ok ? 1.f : 0.f;
    }
    __syncthreads();
    if (sbox[4] > 0.f) {
      float jx1 = sbox[0], jy1 = sbox[1], jx2 = sbox[2], jy2 = sbox[3];
      float areaj = (jx2 - jx1 + 1.f) * (jy2 - jy1 + 1.f);
      #pragma unroll
      for (int j = 0; j < NMSCH; ++j) {
        int idx = j * 1024 + tid;
        if (idx < A9 && sc[j] > -3.0e38f) {
          const float4 bb = *(const float4*)(bx_g + (size_t)idx * 4);
          float iw = fminf(jx2, bb.z) - fmaxf(jx1, bb.x) + 1.f;
          float ih = fminf(jy2, bb.w) - fmaxf(jy1, bb.y) + 1.f;
          iw = fmaxf(iw, 0.f); ih = fmaxf(ih, 0.f);
          float inter = iw * ih;
          float areab = (bb.z - bb.x + 1.f) * (bb.w - bb.y + 1.f);
          float iou = inter / (areaj + areab - inter);
          if (iou > 0.7f) sc[j] = -3.4e38f;
        }
      }
    }
    __syncthreads();
  }
}

// ---------------------------------------------------------------- init scalars
__global__ void init_kernel(float* __restrict__ out, float* __restrict__ gt_max) {
  int t = threadIdx.x;
  if (t < 2) out[BATCH * POST * 5 + t] = 0.f;
  if (t < BATCH * 20) gt_max[t] = 0.f;
}

// ---------------------------------------------------------------- IoU / argmax
__global__ void iou_max_kernel(
    const float* __restrict__ gt_boxes, const int* __restrict__ num_boxes,
    float* __restrict__ max_ov, int* __restrict__ arg_ov,
    float* __restrict__ gt_max)
{
  int gid = blockIdx.x * blockDim.x + threadIdx.x;
  if (gid >= BATCH * A9) return;
  int b = gid / A9, i = gid - b * A9;
  float x1, y1, x2, y2;
  anchor_of(i, x1, y1, x2, y2);
  float area_a = (x2 - x1 + 1.f) * (y2 - y1 + 1.f);
  int nb = num_boxes[b];
  float mo = 0.f; int ag = 0;
  for (int g = 0; g < 20; ++g) {
    float ov = 0.f;
    if (g < nb) {
      const float* gb = gt_boxes + ((size_t)b * 20 + g) * 5;
      float iw = fminf(x2, gb[2]) - fmaxf(x1, gb[0]) + 1.f;
      float ih = fminf(y2, gb[3]) - fmaxf(y1, gb[1]) + 1.f;
      iw = fmaxf(iw, 0.f); ih = fmaxf(ih, 0.f);
      float inter = iw * ih;
      float area_g = (gb[2] - gb[0] + 1.f) * (gb[3] - gb[1] + 1.f);
      ov = inter / (area_a + area_g - inter);
      atomicMax((int*)&gt_max[b * 20 + g], __float_as_int(ov));
    }
    if (ov > mo) { mo = ov; ag = g; }
  }
  max_ov[gid] = mo;
  arg_ov[gid] = ag;
}

// ---------------------------------------------------------------- labels
__global__ void label_kernel(
    const float* __restrict__ gt_boxes, const int* __restrict__ num_boxes,
    const float* __restrict__ im_info, const float* __restrict__ max_ov,
    const float* __restrict__ gt_max, int* __restrict__ labels)
{
  int gid = blockIdx.x * blockDim.x + threadIdx.x;
  if (gid >= BATCH * A9) return;
  int b = gid / A9, i = gid - b * A9;
  float x1, y1, x2, y2;
  anchor_of(i, x1, y1, x2, y2);
  bool inside = (x1 >= 0.f) && (y1 >= 0.f) &&
                (x2 < im_info[1]) && (y2 < im_info[0]);
  int lab = -1;
  if (inside) {
    float mo = max_ov[gid];
    if (mo < 0.3f) lab = 0;
    float area_a = (x2 - x1 + 1.f) * (y2 - y1 + 1.f);
    int nb = num_boxes[b];
    bool best = false;
    for (int g = 0; g < 20; ++g) {
      if (g < nb) {
        const float* gb = gt_boxes + ((size_t)b * 20 + g) * 5;
        float iw = fminf(x2, gb[2]) - fmaxf(x1, gb[0]) + 1.f;
        float ih = fminf(y2, gb[3]) - fmaxf(y1, gb[1]) + 1.f;
        iw = fmaxf(iw, 0.f); ih = fmaxf(ih, 0.f);
        float inter = iw * ih;
        float area_g = (gb[2] - gb[0] + 1.f) * (gb[3] - gb[1] + 1.f);
        float ov = inter / (area_a + area_g - inter);
        float gm = gt_max[b * 20 + g];
        if (gm == 0.f) gm = 1e-5f;
        if (ov == gm) best = true;
      }
    }
    if (best) lab = 1;
    if (mo >= 0.7f) lab = 1;
  }
  labels[gid] = lab;
}

// ---------------------------------------------------------------- subsample
__global__ __launch_bounds__(1024) void subsample_kernel(
    int* __restrict__ labels, int* __restrict__ n_ex)
{
  __shared__ int s[1024];
  __shared__ int sbase;
  const int b = blockIdx.x, tid = threadIdx.x;
  int* lab = labels + (size_t)b * A9;

  // pass 1: foreground, keep lowest-index 128
  if (tid == 0) sbase = 0;
  __syncthreads();
  for (int c = 0; c < NMSCH; ++c) {
    int idx = c * 1024 + tid;
    int flag = (idx < A9 && lab[idx] == 1) ? 1 : 0;
    s[tid] = flag;
    __syncthreads();
    for (int off = 1; off < 1024; off <<= 1) {
      int t = (tid >= off) ? s[tid - off] : 0;
      __syncthreads();
      s[tid] += t;
      __syncthreads();
    }
    int rank = sbase + s[tid] - flag;
    if (flag && rank >= 128) lab[idx] = -1;
    int tot = s[1023];
    __syncthreads();
    if (tid == 0) sbase += tot;
    __syncthreads();
  }
  int fg_total = sbase;
  int fg_cnt = fg_total < 128 ? fg_total : 128;
  int bg_limit = 256 - fg_cnt;
  __syncthreads();

  // pass 2: background, keep lowest-index (256 - fg_cnt)
  if (tid == 0) sbase = 0;
  __syncthreads();
  for (int c = 0; c < NMSCH; ++c) {
    int idx = c * 1024 + tid;
    int flag = (idx < A9 && lab[idx] == 0) ? 1 : 0;
    s[tid] = flag;
    __syncthreads();
    for (int off = 1; off < 1024; off <<= 1) {
      int t = (tid >= off) ? s[tid - off] : 0;
      __syncthreads();
      s[tid] += t;
      __syncthreads();
    }
    int rank = sbase + s[tid] - flag;
    if (flag && rank >= bg_limit) lab[idx] = -1;
    int tot = s[1023];
    __syncthreads();
    if (tid == 0) sbase += tot;
    __syncthreads();
  }
  int bg_total = sbase;
  if (tid == 0) {
    int kept = fg_cnt + (bg_total < bg_limit ? bg_total : bg_limit);
    n_ex[b] = kept > 0 ? kept : 1;
  }
}

// ---------------------------------------------------------------- losses
__global__ void loss_kernel(
    const float* __restrict__ cls_score, const float* __restrict__ bbox_pred,
    const float* __restrict__ gt_boxes, const int* __restrict__ labels,
    const int* __restrict__ arg_ov, const int* __restrict__ n_ex,
    float* __restrict__ out)
{
  int gid = blockIdx.x * blockDim.x + threadIdx.x;
  if (gid >= BATCH * A9) return;
  int lab = labels[gid];
  if (lab < 0) return;
  int b = gid / A9, i = gid - b * A9;
  int p = i / 9, a = i - p * 9;
  float inv = 1.0f / ((float)n_ex[b] * (float)BATCH);

  // masked cross entropy (pair softmax)
  float s0 = cls_score[((size_t)b * 18 + a) * PIX + p];
  float s1 = cls_score[((size_t)b * 18 + 9 + a) * PIX + p];
  float sl = lab ? s1 : s0;
  float so = lab ? s0 : s1;
  float nll = log1pf(expf(so - sl));
  atomicAdd(&out[BATCH * POST * 5 + 0], nll * inv);

  // smooth-L1 box loss for positives
  if (lab == 1) {
    float x1, y1, x2, y2;
    anchor_of(i, x1, y1, x2, y2);
    float aw = x2 - x1 + 1.f, ah = y2 - y1 + 1.f;
    float ax = x1 + 0.5f * aw, ay = y1 + 0.5f * ah;
    const float* gb = gt_boxes + ((size_t)b * 20 + arg_ov[gid]) * 5;
    float gw = gb[2] - gb[0] + 1.f, gh = gb[3] - gb[1] + 1.f;
    float gx = gb[0] + 0.5f * gw, gy = gb[1] + 0.5f * gh;
    float tgt[4] = {(gx - ax) / aw, (gy - ay) / ah, logf(gw / aw), logf(gh / ah)};
    size_t pb = ((size_t)b * 36 + a * 4) * PIX + p;
    float lsum = 0.f;
    #pragma unroll
    for (int d = 0; d < 4; ++d) {
      float diff = bbox_pred[pb + (size_t)d * PIX] - tgt[d];
      float ad = fabsf(diff);
      lsum += (ad < 1.f / 9.f) ? 4.5f * diff * diff : (ad - 1.f / 18.f);
    }
    atomicAdd(&out[BATCH * POST * 5 + 1], lsum * inv);
  }
}

// ---------------------------------------------------------------- launch
extern "C" void kernel_launch(void* const* d_in, const int* in_sizes, int n_in,
                              void* d_out, int out_size, void* d_ws, size_t ws_size,
                              hipStream_t stream) {
  const float* base_feat = (const float*)d_in[0];
  const float* im_info   = (const float*)d_in[1];
  const float* gt_boxes  = (const float*)d_in[2];
  const int*   num_boxes = (const int*)d_in[3];
  const float* conv_w    = (const float*)d_in[4];
  const float* conv_b    = (const float*)d_in[5];
  const float* cls_w     = (const float*)d_in[6];
  const float* cls_b     = (const float*)d_in[7];
  const float* bbox_w    = (const float*)d_in[8];
  const float* bbox_b    = (const float*)d_in[9];
  float* out = (float*)d_out;

  char* ws = (char*)d_ws;
  size_t off = 0;
  auto alloc = [&](size_t bytes) -> void* {
    void* p = ws + off;
    off = (off + bytes + 255) & ~(size_t)255;
    return p;
  };
  half_t* wtt = (half_t*)alloc((size_t)KDIM * NOUT * sizeof(half_t));
  char* xregion = ws + off;
  float* x = (float*)alloc((size_t)BATCH * NOUT * PIX * sizeof(float));
  float* cls_score = (float*)alloc((size_t)BATCH * 18 * PIX * sizeof(float));
  float* bbox_pred = (float*)alloc((size_t)BATCH * 36 * PIX * sizeof(float));

  // buffers that come alive only after x is dead -> alias over x's region
  size_t aoff = 0;
  auto alloc2 = [&](size_t bytes) -> void* {
    void* p = xregion + aoff;
    aoff = (aoff + bytes + 255) & ~(size_t)255;
    return p;
  };
  float* scores = (float*)alloc2((size_t)BATCH * A9 * sizeof(float));
  float* props  = (float*)alloc2((size_t)BATCH * A9 * 4 * sizeof(float));
  float* max_ov = (float*)alloc2((size_t)BATCH * A9 * sizeof(float));
  int*   arg_ov = (int*)  alloc2((size_t)BATCH * A9 * sizeof(int));
  int*   labels = (int*)  alloc2((size_t)BATCH * A9 * sizeof(int));
  float* gt_max = (float*)alloc2((size_t)BATCH * 20 * sizeof(float));
  int*   n_ex   = (int*)  alloc2((size_t)BATCH * sizeof(int));

  // 1. weights -> f16, column-major [n][k']
  {
    int n = KDIM * NOUT;
    prep_wtt<<<(n + 255) / 256, 256, 0, stream>>>(conv_w, wtt, n);
  }
  // 2. conv3x3 + bias + relu via WMMA implicit GEMM
  {
    dim3 grid((MROWS + 127) / 128, NOUT / 128);
    conv3x3_wmma<<<grid, 256, 0, stream>>>(base_feat, wtt, conv_b, x);
  }
  // 3. 1x1 heads
  {
    int n = 54 * MROWS;
    head1x1<<<(n + 255) / 256, 256, 0, stream>>>(x, cls_w, cls_b, bbox_w, bbox_b,
                                                 cls_score, bbox_pred);
  }
  // 4. anchor decode + clip + fg score
  {
    int n = BATCH * A9;
    proposal_decode<<<(n + 255) / 256, 256, 0, stream>>>(cls_score, bbox_pred,
                                                         im_info, scores, props);
  }
  // 5. zero loss scalars + gt_max (aliased region is now live)
  init_kernel<<<1, 256, 0, stream>>>(out, gt_max);
  // 6. greedy NMS per image -> rois in d_out
  nms_kernel<<<BATCH, 1024, 0, stream>>>(scores, props, out);
  // 7. anchor targets
  {
    int n = BATCH * A9;
    int g = (n + 255) / 256;
    iou_max_kernel<<<g, 256, 0, stream>>>(gt_boxes, num_boxes, max_ov, arg_ov, gt_max);
    label_kernel<<<g, 256, 0, stream>>>(gt_boxes, num_boxes, im_info, max_ov,
                                        gt_max, labels);
    subsample_kernel<<<BATCH, 1024, 0, stream>>>(labels, n_ex);
    loss_kernel<<<g, 256, 0, stream>>>(cls_score, bbox_pred, gt_boxes, labels,
                                       arg_ov, n_ex, out);
  }
}
